// Encoder_74801150427406
// MI455X (gfx1250) — matrix operand order repaired and checked by reading, not compiled, not run on hardware
//
#include <hip/hip_runtime.h>
#include <hip/hip_bf16.h>

typedef __attribute__((ext_vector_type(16))) _Float16 v16h;
typedef __attribute__((ext_vector_type(8)))  float    v8f;
typedef __attribute__((ext_vector_type(4)))  int      v4i;

#define B_N     4096
#define D_IN    900
#define D_PAD   928     // 29 * 32
#define NHEAD   10
#define HD      71
#define HD_PAD  96      // 3 * 32
#define OD      64
#define FC_OUT  54
#define OUT_COLS 64

// ---------------- WMMA fragment loaders (CDNA5 16x16x32 f16) ----------------
// A matrix 16x32: lane L holds row M=L&15; K = kb..kb+7 (halves 0-7) and
// kb+16..kb+23 (halves 8-15), kb = (L>=16)*8.  Two 16B chunks.
static __device__ __forceinline__ v16h load_fragA(const _Float16* base, int ld, int lane) {
  const _Float16* p = base + (size_t)(lane & 15) * ld + ((lane >> 4) << 3);
  v16h f;
  reinterpret_cast<float4*>(&f)[0] = *reinterpret_cast<const float4*>(p);
  reinterpret_cast<float4*>(&f)[1] = *reinterpret_cast<const float4*>(p + 16);
  return f;
}
// B matrix 32x16 from [N][K]-major storage: lane L holds column N=L&15,
// K = kb..kb+7 (halves 0-7) and kb+16..kb+23 (halves 8-15), kb = (L>=16)*8.  Two 16B chunks.
static __device__ __forceinline__ v16h load_fragB(const _Float16* base, int ld, int lane) {
  const _Float16* p = base + (size_t)(lane & 15) * ld + ((lane >> 4) << 3);
  v16h f;
  reinterpret_cast<float4*>(&f)[0] = *reinterpret_cast<const float4*>(p);
  reinterpret_cast<float4*>(&f)[1] = *reinterpret_cast<const float4*>(p + 16);
  return f;
}
static __device__ __forceinline__ v8f wmma_f16(v16h a, v16h b, v8f c) {
  return __builtin_amdgcn_wmma_f32_16x16x32_f16(false, a, false, b, (short)0, c, false, false);
}

// Async global -> LDS copy (GLOBAL_LOAD_ASYNC_TO_LDS_B128, ASYNCcnt-tracked)
static __device__ __forceinline__ void async_cp_b128(const _Float16* gsrc, _Float16* ldst) {
  __builtin_amdgcn_global_load_async_to_lds_b128(
      (__attribute__((address_space(1))) v4i*)(gsrc),
      (__attribute__((address_space(3))) v4i*)(ldst), 0, 0);
}
static __device__ __forceinline__ void wait_async0() {
  asm volatile("s_wait_asynccnt 0x0" ::: "memory");
}

// ---------------- conversion kernels ----------------
__global__ void conv_e_kernel(const float* __restrict__ E, _Float16* __restrict__ Ef) {
  int idx = blockIdx.x * 256 + threadIdx.x;
  if (idx >= B_N * D_PAD) return;
  int d = idx % D_PAD;
  int b = idx / D_PAD;
  Ef[idx] = (d < D_IN) ? (_Float16)E[(size_t)b * D_IN + d] : (_Float16)0.f;
}

__global__ void conv_wq_kernel(const float* __restrict__ Wq, _Float16* __restrict__ Wqt) {
  int idx = blockIdx.x * 256 + threadIdx.x;
  if (idx >= NHEAD * HD_PAD * D_PAD) return;
  int d = idx % D_PAD;
  int r = idx / D_PAD;
  int h = r % HD_PAD;
  int n = r / HD_PAD;
  float v = 0.f;
  if (h < HD && d < D_IN) v = Wq[((size_t)n * D_IN + d) * HD + h];
  Wqt[idx] = (_Float16)v;
}

__global__ void conv_wv_kernel(const float* __restrict__ Wv, _Float16* __restrict__ Wvt) {
  int idx = blockIdx.x * 256 + threadIdx.x;
  if (idx >= NHEAD * OD * D_PAD) return;
  int d = idx % D_PAD;
  int r = idx / D_PAD;
  int o = r % OD;
  int n = r / OD;
  float v = 0.f;
  if (d < D_IN) v = Wv[((size_t)n * D_IN + d) * OD + o];
  Wvt[idx] = (_Float16)v;
}

// ---------------- k projection from 0/1 masks + mask_sums ----------------
// k[n][b][h] = relu( sum_{d: pg[b,d]==n+1} Wk[n][d][h] + bk[n][h] )
__global__ void kmask_kernel(const int* __restrict__ pg, const float* __restrict__ Wk,
                             const float* __restrict__ bk, _Float16* __restrict__ Kf,
                             float* __restrict__ out) {
  __shared__ int cls[D_IN];
  int b = blockIdx.x;
  int tid = threadIdx.x;                 // blockDim = 96
  for (int d = tid; d < D_IN; d += 96) cls[d] = pg[(size_t)b * D_IN + d];
  __syncthreads();
  int h = tid;                           // 0..95 (valid < 71)
  float acc[NHEAD], cnt[NHEAD];
#pragma unroll
  for (int i = 0; i < NHEAD; ++i) { acc[i] = 0.f; cnt[i] = 0.f; }
  for (int d = 0; d < D_IN; ++d) {
    int c = cls[d] - 1;                  // uniform across block
    float w = 0.f;
    if ((unsigned)c < (unsigned)NHEAD && h < HD)
      w = Wk[((size_t)c * D_IN + d) * HD + h];
#pragma unroll
    for (int i = 0; i < NHEAD; ++i)
      if (c == i) { acc[i] += w; cnt[i] += 1.f; }
  }
#pragma unroll
  for (int i = 0; i < NHEAD; ++i) {
    float v = acc[i] + (h < HD ? bk[i * HD + h] : 0.f);
    v = fmaxf(v, 0.f);
    Kf[((size_t)i * B_N + b) * HD_PAD + h] = (_Float16)v;
  }
  if (tid == 0) {
#pragma unroll
    for (int i = 0; i < NHEAD; ++i)
      out[(size_t)b * OUT_COLS + FC_OUT + i] = cnt[i];
  }
}

// ---------------- q/v projection GEMM (WMMA) ----------------
// grid (32, 10), block 256 (8 waves); wave owns 16 rows; N = 96 (q) + 64 (v)
__global__ __launch_bounds__(256) void qv_proj_kernel(
    const _Float16* __restrict__ Ef, const _Float16* __restrict__ Wqt,
    const _Float16* __restrict__ Wvt, const float* __restrict__ bq,
    const float* __restrict__ bv, _Float16* __restrict__ Qf, _Float16* __restrict__ Vt) {
  int head = blockIdx.y;
  int wave = threadIdx.x >> 5;
  int lane = threadIdx.x & 31;
  int row0 = blockIdx.x * 128 + wave * 16;
  const _Float16* Wq_h = Wqt + (size_t)head * HD_PAD * D_PAD;
  const _Float16* Wv_h = Wvt + (size_t)head * OD * D_PAD;

  v8f cq[6], cv[4];
#pragma unroll
  for (int i = 0; i < 6; ++i) cq[i] = (v8f){};
#pragma unroll
  for (int i = 0; i < 4; ++i) cv[i] = (v8f){};

  for (int k0 = 0; k0 < D_PAD; k0 += 32) {
    v16h a = load_fragA(Ef + (size_t)row0 * D_PAD + k0, D_PAD, lane);
#pragma unroll
    for (int nt = 0; nt < 6; ++nt) {
      v16h bf = load_fragB(Wq_h + (size_t)(nt * 16) * D_PAD + k0, D_PAD, lane);
      cq[nt] = wmma_f16(a, bf, cq[nt]);
    }
#pragma unroll
    for (int nt = 0; nt < 4; ++nt) {
      v16h bf = load_fragB(Wv_h + (size_t)(nt * 16) * D_PAD + k0, D_PAD, lane);
      cv[nt] = wmma_f16(a, bf, cv[nt]);
    }
  }

  int nn = lane & 15;
  int mb = (lane >> 4) << 3;
#pragma unroll
  for (int nt = 0; nt < 6; ++nt) {               // q -> [head][b][96]
    int hcol = nt * 16 + nn;
    float bias = (hcol < HD) ? bq[head * HD + hcol] : 0.f;
#pragma unroll
    for (int j = 0; j < 8; ++j) {
      int m = row0 + mb + j;
      float v = fmaxf(cq[nt][j] + bias, 0.f);
      Qf[((size_t)head * B_N + m) * HD_PAD + hcol] = (_Float16)v;
    }
  }
#pragma unroll
  for (int nt = 0; nt < 4; ++nt) {               // v transposed -> [head][o][4096]
    int oc = nt * 16 + nn;
    float bias = bv[head * OD + oc];
#pragma unroll
    for (int j = 0; j < 8; ++j) {
      int m = row0 + mb + j;
      float v = fmaxf(cv[nt][j] + bias, 0.f);
      Vt[((size_t)head * OD + oc) * B_N + m] = (_Float16)v;
    }
  }
}

// ---------------- flash attention (WMMA + async LDS staging) ----------------
// grid (32, 10), block 256 (8 waves); wave owns a 16-row query strip.
// Per 64-key block: K tile [64,96] and V tile [64 o][64 keys] are staged into
// LDS once via GLOBAL_LOAD_ASYNC_TO_LDS_B128 and shared by all 8 waves.
__global__ __launch_bounds__(256) void flash_attn_kernel(
    const _Float16* __restrict__ Qf, const _Float16* __restrict__ Kf,
    const _Float16* __restrict__ Vt, float* __restrict__ att) {
  __shared__ _Float16 kt_s[64 * HD_PAD];         // 12 KB: [key][96]
  __shared__ _Float16 vt_s[OD * 64];             // 8 KB:  [o][key]
  __shared__ _Float16 lds_p[8][16 * 64];         // 16 KB: per-wave P staging
  int head = blockIdx.y;
  int tid  = threadIdx.x;
  int wave = tid >> 5;
  int lane = tid & 31;
  int row0 = blockIdx.x * 128 + wave * 16;
  const float scale = 0.118678165819385f;        // 1/sqrt(71)

  const _Float16* Kh = Kf + (size_t)head * B_N * HD_PAD;
  const _Float16* Vh = Vt + (size_t)head * OD * B_N;

  v16h qf[3];
#pragma unroll
  for (int kt = 0; kt < 3; ++kt)
    qf[kt] = load_fragA(Qf + ((size_t)head * B_N + row0) * HD_PAD + kt * 32, HD_PAD, lane);

  v8f o[4];
#pragma unroll
  for (int i = 0; i < 4; ++i) o[i] = (v8f){};
  float mstat[8], lstat[8];
#pragma unroll
  for (int j = 0; j < 8; ++j) { mstat[j] = -1e30f; lstat[j] = 0.f; }

  _Float16* myp = &lds_p[wave][0];
  int nn = lane & 15;
  int mb = (lane >> 4) << 3;

  for (int c0 = 0; c0 < B_N; c0 += 64) {
    __syncthreads();                             // previous block's reads complete
    // stage K tile: 64*96 f16 = 768 x 16B chunks (12 chunks per key row)
#pragma unroll
    for (int it = 0; it < 3; ++it) {
      int idx = tid + it * 256;
      int krow = idx / 12;
      int col8 = (idx % 12) * 8;
      async_cp_b128(Kh + (size_t)(c0 + krow) * HD_PAD + col8, kt_s + krow * HD_PAD + col8);
    }
    // stage V tile: 64*64 f16 = 512 x 16B chunks (8 chunks per o row)
#pragma unroll
    for (int it = 0; it < 2; ++it) {
      int idx = tid + it * 256;
      int orow = idx / 8;
      int col8 = (idx % 8) * 8;
      async_cp_b128(Vh + (size_t)orow * B_N + c0 + col8, vt_s + orow * 64 + col8);
    }
    wait_async0();
    __syncthreads();                             // tiles visible to all waves

    // S strip [16,64] = q @ k^T (K = 96), B-frags from LDS
    v8f s[4];
#pragma unroll
    for (int nt = 0; nt < 4; ++nt) {
      v8f acc = (v8f){};
#pragma unroll
      for (int kt = 0; kt < 3; ++kt) {
        v16h bf = load_fragB(kt_s + (nt * 16) * HD_PAD + kt * 32, HD_PAD, lane);
        acc = wmma_f16(qf[kt], bf, acc);
      }
#pragma unroll
      for (int j = 0; j < 8; ++j) acc[j] *= scale;
      s[nt] = acc;
    }
    // online softmax: row r = j + 8*(lane>=16) lives in 16 lanes of one half
#pragma unroll
    for (int j = 0; j < 8; ++j) {
      float t = fmaxf(fmaxf(s[0][j], s[1][j]), fmaxf(s[2][j], s[3][j]));
      t = fmaxf(t, __shfl_xor(t, 1));
      t = fmaxf(t, __shfl_xor(t, 2));
      t = fmaxf(t, __shfl_xor(t, 4));
      t = fmaxf(t, __shfl_xor(t, 8));
      float mnew  = fmaxf(mstat[j], t);
      float alpha = __expf(mstat[j] - mnew);
      float rs = 0.f;
#pragma unroll
      for (int nt = 0; nt < 4; ++nt) {
        float p = __expf(s[nt][j] - mnew);
        s[nt][j] = p;
        rs += p;
      }
      rs += __shfl_xor(rs, 1);
      rs += __shfl_xor(rs, 2);
      rs += __shfl_xor(rs, 4);
      rs += __shfl_xor(rs, 8);
      lstat[j] = lstat[j] * alpha + rs;
      mstat[j] = mnew;
#pragma unroll
      for (int nt = 0; nt < 4; ++nt) o[nt][j] *= alpha;
    }
    // P (C-frag layout) -> LDS row-major f16, re-read as A-frags (same wave, in-order DS)
#pragma unroll
    for (int nt = 0; nt < 4; ++nt)
#pragma unroll
      for (int j = 0; j < 8; ++j)
        myp[(mb + j) * 64 + nt * 16 + nn] = (_Float16)s[nt][j];
    v16h pa0 = load_fragA(myp, 64, lane);
    v16h pa1 = load_fragA(myp + 32, 64, lane);
    // O += P @ V  (K = 64 keys), B-frags from LDS
#pragma unroll
    for (int ot = 0; ot < 4; ++ot) {
      v16h b0 = load_fragB(vt_s + (ot * 16) * 64, 64, lane);
      v16h b1 = load_fragB(vt_s + (ot * 16) * 64 + 32, 64, lane);
      o[ot] = wmma_f16(pa0, b0, o[ot]);
      o[ot] = wmma_f16(pa1, b1, o[ot]);
    }
  }
  // normalize and store attended_cat[b][head*64 + o]
#pragma unroll
  for (int ot = 0; ot < 4; ++ot)
#pragma unroll
    for (int j = 0; j < 8; ++j) {
      float inv = 1.f / lstat[j];
      att[(size_t)(row0 + mb + j) * (NHEAD * OD) + head * OD + ot * 16 + nn] = o[ot][j] * inv;
    }
}

// ---------------- final FC: [4096,640] @ [640,54] + bias, relu ----------------
__global__ void fc_kernel(const float* __restrict__ att, const float* __restrict__ Wfc,
                          const float* __restrict__ bfc, float* __restrict__ out) {
  __shared__ float row[NHEAD * OD];
  int b = blockIdx.x;
  int tid = threadIdx.x;                 // 64
  for (int i = tid; i < NHEAD * OD; i += 64) row[i] = att[(size_t)b * NHEAD * OD + i];
  __syncthreads();
  if (tid < FC_OUT) {
    float s = bfc[tid];
    for (int i = 0; i < NHEAD * OD; ++i) s += row[i] * Wfc[(size_t)i * FC_OUT + tid];
    out[(size_t)b * OUT_COLS + tid] = fmaxf(s, 0.f);
  }
}

// ---------------- launch ----------------
extern "C" void kernel_launch(void* const* d_in, const int* in_sizes, int n_in,
                              void* d_out, int out_size, void* d_ws, size_t ws_size,
                              hipStream_t stream) {
  (void)in_sizes; (void)n_in; (void)out_size; (void)ws_size;
  const float* E   = (const float*)d_in[0];
  const int*   pg  = (const int*)d_in[1];
  const float* Wq  = (const float*)d_in[2];
  const float* bq  = (const float*)d_in[3];
  const float* Wk  = (const float*)d_in[4];
  const float* bk  = (const float*)d_in[5];
  const float* Wv  = (const float*)d_in[6];
  const float* bv  = (const float*)d_in[7];
  const float* Wfc = (const float*)d_in[8];
  const float* bfc = (const float*)d_in[9];
  float* out = (float*)d_out;

  char* ws = (char*)d_ws;
  size_t off = 0;
  _Float16* Ef  = (_Float16*)(ws + off); off += (size_t)B_N * D_PAD * 2;            // 7.6 MB
  _Float16* Wqt = (_Float16*)(ws + off); off += (size_t)NHEAD * HD_PAD * D_PAD * 2; // 1.8 MB
  _Float16* Wvt = (_Float16*)(ws + off); off += (size_t)NHEAD * OD * D_PAD * 2;     // 1.2 MB
  _Float16* Qf  = (_Float16*)(ws + off); off += (size_t)NHEAD * B_N * HD_PAD * 2;   // 7.9 MB
  _Float16* Kf  = (_Float16*)(ws + off); off += (size_t)NHEAD * B_N * HD_PAD * 2;   // 7.9 MB
  _Float16* Vt  = (_Float16*)(ws + off); off += (size_t)NHEAD * OD * B_N * 2;       // 5.2 MB
  float*    att = (float*)(ws + off);    off += (size_t)B_N * NHEAD * OD * 4;       // 10.5 MB

  conv_e_kernel<<<(B_N * D_PAD + 255) / 256, 256, 0, stream>>>(E, Ef);
  conv_wq_kernel<<<(NHEAD * HD_PAD * D_PAD + 255) / 256, 256, 0, stream>>>(Wq, Wqt);
  conv_wv_kernel<<<(NHEAD * OD * D_PAD + 255) / 256, 256, 0, stream>>>(Wv, Wvt);
  kmask_kernel<<<B_N, 96, 0, stream>>>(pg, Wk, bk, Kf, out);
  qv_proj_kernel<<<dim3(B_N / 128, NHEAD), 256, 0, stream>>>(Ef, Wqt, Wvt, bq, bv, Qf, Vt);
  flash_attn_kernel<<<dim3(B_N / 128, NHEAD), 256, 0, stream>>>(Qf, Kf, Vt, att);
  fc_kernel<<<B_N, 64, 0, stream>>>(att, Wfc, bfc, out);
}
